// TestLSTM_74560632259269
// MI455X (gfx1250) — matrix operand order repaired and checked by reading, not compiled, not run on hardware
//
#include <hip/hip_runtime.h>
#include <math.h>

#define VOCAB 50257
#define EMB 1024
#define HID 1024
#define SEQ 2048
#define GATES 4096          // 4*HID

typedef __attribute__((ext_vector_type(16))) __bf16         v16bf;
typedef __attribute__((ext_vector_type(8)))  float          v8f;
typedef __attribute__((ext_vector_type(4)))  unsigned int   u32x4;
typedef __attribute__((ext_vector_type(4)))  float          f32x4;
typedef __attribute__((ext_vector_type(4)))  unsigned short u16x4;

union BFrag { v16bf bf; u32x4 q[2]; };

__device__ __forceinline__ unsigned short f2bf(float f) {
    unsigned int u = __float_as_uint(f);
    unsigned int r = u + 0x7FFFu + ((u >> 16) & 1u);   // round-to-nearest-even
    return (unsigned short)(r >> 16);
}
__device__ __forceinline__ float bf2f_lo(unsigned int packed) {
    return __uint_as_float(packed << 16);
}
__device__ __forceinline__ float bf2f_hi(unsigned int packed) {
    return __uint_as_float(packed & 0xFFFF0000u);
}
__device__ __forceinline__ float sigmoidf_(float x) {
    return 1.0f / (1.0f + __expf(-x));
}

// ---------------------------------------------------------------------------
// Kernel 1: convert W_ih / W_hh to bf16 (4 elems/thread, b128 in / b64 out),
// compute bias sum.
// ---------------------------------------------------------------------------
__global__ void prep_weights_kernel(const float* __restrict__ wih,
                                    const float* __restrict__ whh,
                                    const float* __restrict__ bih,
                                    const float* __restrict__ bhh,
                                    unsigned short* __restrict__ wb,
                                    unsigned short* __restrict__ whb,
                                    float* __restrict__ bsum) {
    int i4 = blockIdx.x * blockDim.x + threadIdx.x;     // covers (4096*1024)/4
    long i = (long)i4 * 4;
    f32x4 a = *(const f32x4*)(wih + i);
    f32x4 b = *(const f32x4*)(whh + i);
    u16x4 pa = { f2bf(a.x), f2bf(a.y), f2bf(a.z), f2bf(a.w) };
    u16x4 pb = { f2bf(b.x), f2bf(b.y), f2bf(b.z), f2bf(b.w) };
    *(u16x4*)(wb + i)  = pa;
    *(u16x4*)(whb + i) = pb;
    if (i < GATES) {
        f32x4 x = *(const f32x4*)(bih + i);
        f32x4 y = *(const f32x4*)(bhh + i);
        *(f32x4*)(bsum + i) = x + y;
    }
}

// ---------------------------------------------------------------------------
// Kernel 2: embedding gather + downcast, 4 elems/thread.
// ---------------------------------------------------------------------------
__global__ void gather_embed_kernel(const int* __restrict__ sentence,
                                    const float* __restrict__ emb,
                                    unsigned short* __restrict__ xb) {
    int t = blockIdx.x;
    const float* src = emb + (long)sentence[t] * EMB;
    unsigned short* dst = xb + (long)t * EMB;
    int e = threadIdx.x * 4;                            // blockDim 256 -> 1024 elems
    f32x4 v = *(const f32x4*)(src + e);
    u16x4 p = { f2bf(v.x), f2bf(v.y), f2bf(v.z), f2bf(v.w) };
    *(u16x4*)(dst + e) = p;
}

// ---------------------------------------------------------------------------
// Kernel 3: gx[2048][4096] = xb @ Wb^T + bsum  via v_wmma_f32_16x16x32_bf16.
// One 16x16 C tile per wave; K loop over 1024 in steps of 32.
// ---------------------------------------------------------------------------
__global__ void gx_gemm_kernel(const unsigned short* __restrict__ xb,
                               const unsigned short* __restrict__ wb,
                               const float* __restrict__ bsum,
                               float* __restrict__ gx) {
    const int wid    = threadIdx.x >> 5;
    const int lane   = threadIdx.x & 31;
    const int tileId = blockIdx.x * 8 + wid;            // 32768 tiles total
    const int mt = tileId & 127;                        // 128 M tiles (SEQ/16)
    const int nt = tileId >> 7;                         // 256 N tiles (GATES/16)
    const int m0 = mt * 16;
    const int n0 = nt * 16;

    const int r  = lane & 15;
    const int kh = lane >> 4;

    const unsigned short* aBase = xb + (long)(m0 + r) * EMB + kh * 16;
    const unsigned short* bBase = wb + (long)(n0 + r) * EMB + kh * 16;

    const float bv = bsum[n0 + r];
    v8f acc;
    #pragma unroll
    for (int v = 0; v < 8; ++v) acc[v] = bv;

    #pragma unroll 4
    for (int k0 = 0; k0 < EMB; k0 += 32) {
        BFrag a, b;
        a.q[0] = *(const u32x4*)(aBase + k0);
        a.q[1] = *(const u32x4*)(aBase + k0 + 8);
        b.q[0] = *(const u32x4*)(bBase + k0);
        b.q[1] = *(const u32x4*)(bBase + k0 + 8);
        acc = __builtin_amdgcn_wmma_f32_16x16x32_bf16(
            /*neg_a=*/false, a.bf, /*neg_b=*/false, b.bf,
            /*c_mod=*/(short)0, acc, /*reuse_a=*/false, /*reuse_b=*/false);
    }

    float* outp = gx + (long)(m0 + kh * 8) * GATES + n0 + r;
    #pragma unroll
    for (int v = 0; v < 8; ++v) outp[(long)v * GATES] = acc[v];
}

// ---------------------------------------------------------------------------
// Kernel 4: persistent LSTM recurrence.
// 64 blocks x 256 threads. Block w owns hidden indices j = w*16 .. w*16+15 and
// the 64 gate rows {g*1024 + w*16 + jj} of W_hh, pinned in LDS as bf16
// (128 KB). Per step: stage h(t-1) (b128 loads), 64 dots with b128 DS reads,
// gate math, write h(t) to d_out, device-wide atomic barrier.
// ---------------------------------------------------------------------------
__global__ void lstm_recurrence_kernel(const unsigned short* __restrict__ whb,
                                       const float* __restrict__ gx,
                                       float* __restrict__ out,
                                       unsigned int* __restrict__ counter) {
    extern __shared__ char smem[];
    unsigned short* w_s   = (unsigned short*)smem;                 // 131072 B
    float*          h_s   = (float*)(smem + 64 * 1024 * 2);        // 4096 B
    float*          dot_s = (float*)(smem + 64 * 1024 * 2 + 4096); // 256 B
    __shared__ float c_s[16];

    const int w    = blockIdx.x;          // 0..63
    const int tid  = threadIdx.x;         // 0..255
    const int rr   = tid >> 2;            // local row 0..63  (gate = rr>>4, jj = rr&15)
    const int part = tid & 3;             // quarter of the K range

    // --- preload W_hh slice into LDS (bf16, b128 copies) -------------------
    {
        const int grow = (rr >> 4) * HID + w * 16 + (rr & 15);
        const unsigned short* src = whb + (long)grow * HID + part * 256;
        unsigned short* dst = w_s + rr * 1024 + part * 256;
        #pragma unroll
        for (int k = 0; k < 256; k += 8)
            *(u32x4*)(dst + k) = *(const u32x4*)(src + k);
    }
    if (tid < 16) c_s[tid] = 0.0f;
    __syncthreads();

    const unsigned int nblocks = gridDim.x;

    for (int t = 0; t < SEQ; ++t) {
        // stage h(t-1) into LDS: 1024 floats, 1 float4 per thread
        {
            f32x4* hd = (f32x4*)(h_s) + tid;
            if (t == 0) {
                f32x4 z = {0.f, 0.f, 0.f, 0.f};
                *hd = z;
            } else {
                const f32x4* hprev = (const f32x4*)(out + (long)(t - 1) * HID);
                *hd = hprev[tid];
            }
        }
        __syncthreads();

        // 64 dot products, 4 lanes per row, 256 elements each, b128 DS reads
        {
            const u32x4* wr  = (const u32x4*)(w_s + rr * 1024 + part * 256);
            const f32x4* hp4 = (const f32x4*)(h_s + part * 256);
            float sum = 0.0f;
            #pragma unroll 4
            for (int k8 = 0; k8 < 32; ++k8) {
                u32x4 ww = wr[k8];
                f32x4 h0 = hp4[2 * k8];
                f32x4 h1 = hp4[2 * k8 + 1];
                sum = fmaf(bf2f_lo(ww.x), h0.x, sum);
                sum = fmaf(bf2f_hi(ww.x), h0.y, sum);
                sum = fmaf(bf2f_lo(ww.y), h0.z, sum);
                sum = fmaf(bf2f_hi(ww.y), h0.w, sum);
                sum = fmaf(bf2f_lo(ww.z), h1.x, sum);
                sum = fmaf(bf2f_hi(ww.z), h1.y, sum);
                sum = fmaf(bf2f_lo(ww.w), h1.z, sum);
                sum = fmaf(bf2f_hi(ww.w), h1.w, sum);
            }
            sum += __shfl_xor(sum, 1);
            sum += __shfl_xor(sum, 2);
            if (part == 0) dot_s[rr] = sum;
        }
        __syncthreads();

        // gate math for the 16 local hidden indices
        if (tid < 16) {
            const int j = tid;
            const float* gxr = gx + (long)t * GATES + w * 16 + j;
            float gi = gxr[0]        + dot_s[j];
            float gf = gxr[HID]      + dot_s[16 + j];
            float gg = gxr[2 * HID]  + dot_s[32 + j];
            float go = gxr[3 * HID]  + dot_s[48 + j];
            float i_ = sigmoidf_(gi);
            float f_ = sigmoidf_(gf);
            float g_ = tanhf(gg);
            float o_ = sigmoidf_(go);
            float c  = f_ * c_s[j] + i_ * g_;
            c_s[j] = c;
            out[(long)t * HID + w * 16 + j] = o_ * tanhf(c);
        }

        // device-wide barrier: publish h(t), wait for all 64 blocks
        __threadfence();
        __syncthreads();
        if (tid == 0) {
            atomicAdd(counter, 1u);
            const unsigned int target = nblocks * (unsigned int)(t + 1);
            while (atomicAdd(counter, 0u) < target) {
                __builtin_amdgcn_s_sleep(1);
            }
        }
        __syncthreads();
        __threadfence();
    }
}

// ---------------------------------------------------------------------------
// Host launcher
// ---------------------------------------------------------------------------
extern "C" void kernel_launch(void* const* d_in, const int* in_sizes, int n_in,
                              void* d_out, int out_size, void* d_ws, size_t ws_size,
                              hipStream_t stream) {
    const int*   sentence = (const int*)d_in[0];
    const float* emb      = (const float*)d_in[1];
    const float* W_ih     = (const float*)d_in[2];
    const float* W_hh     = (const float*)d_in[3];
    const float* b_ih     = (const float*)d_in[4];
    const float* b_hh     = (const float*)d_in[5];
    float* out = (float*)d_out;

    // workspace layout (bytes)
    char* ws = (char*)d_ws;
    unsigned short* wb   = (unsigned short*)(ws);                       // 8 MB  W_ih bf16
    unsigned short* whb  = (unsigned short*)(ws + (8u << 20));          // 8 MB  W_hh bf16
    float*          bsum = (float*)(ws + (16u << 20));                  // 16 KB
    unsigned short* xb   = (unsigned short*)(ws + (16u << 20) + 65536); // 4 MB  x bf16
    float*          gx   = (float*)(ws + (21u << 20));                  // 32 MB gx
    unsigned int*   ctr  = (unsigned int*)(ws + (54u << 20));           // barrier counter

    // 1. weight downcast + bias sum   (4096*1024 elements, 4 per thread)
    prep_weights_kernel<<<(GATES * EMB) / (256 * 4), 256, 0, stream>>>(
        W_ih, W_hh, b_ih, b_hh, wb, whb, bsum);

    // 2. embedding gather + downcast
    gather_embed_kernel<<<SEQ, 256, 0, stream>>>(sentence, emb, xb);

    // 3. gx = x @ W_ih^T + bias  (WMMA bf16, 8 waves/block, 1 tile/wave)
    gx_gemm_kernel<<<(128 * 256) / 8, 256, 0, stream>>>(xb, wb, bsum, gx);

    // 4. reset barrier counter, run persistent recurrence
    hipMemsetAsync(ctr, 0, 256, stream);

    const int smemBytes = 64 * 1024 * 2 + 4096 + 256;   // 135424 B dynamic LDS
    hipFuncSetAttribute((const void*)lstm_recurrence_kernel,
                        hipFuncAttributeMaxDynamicSharedMemorySize, smemBytes);
    lstm_recurrence_kernel<<<64, 256, smemBytes, stream>>>(whb, gx, out, ctr);
}
